// diffRNN_6313601925761
// MI455X (gfx1250) — compile-verified, matrix operand order checked by reading, not run on hardware
//
#include <hip/hip_runtime.h>
#include <hip/hip_bf16.h>

// ---------------------------------------------------------------------------
// Problem constants (from reference)
// ---------------------------------------------------------------------------
static constexpr int B  = 256;
static constexpr int V  = 50;
static constexpr int Cc = 40;
static constexpr int D  = 256;
static constexpr int H  = 256;
static constexpr int NH = 8;
static constexpr int T  = 1000;

// d_out layout (floats): res[B,V,2] | resg[B,V,2] | res[:,-1][B,2] | resg[:,-1][B,2]
//                        | normal_noise[B,V,D] | predicted_noise[B,V,D]
static constexpr long OFF_RESG = (long)B * V * 2;            // 25600
static constexpr long OFF_RL   = OFF_RESG + (long)B * V * 2; // 51200
static constexpr long OFF_RGL  = OFF_RL + (long)B * 2;       // 51712
static constexpr long OFF_NN   = OFF_RGL + (long)B * 2;      // 52224
static constexpr long OFF_PN   = OFF_NN + (long)B * V * D;   // 3329024

typedef _Float16 h8   __attribute__((ext_vector_type(8)));
typedef _Float16 v16h __attribute__((ext_vector_type(16)));
typedef float    v8f  __attribute__((ext_vector_type(8)));

// gfx1250 async global->LDS path (guarded so either toolchain compiles).
// Probe-verified signature: (global int4* src, shared int4* dst, imm off, imm cpol)
#if defined(__AMDGCN__) && __has_builtin(__builtin_amdgcn_global_load_async_to_lds_b128) && \
    __has_builtin(__builtin_amdgcn_s_wait_asynccnt)
#define USE_ASYNC_LDS 1
typedef int v4i __attribute__((ext_vector_type(4)));
typedef __attribute__((address_space(1))) v4i glob_v4i_t;
typedef __attribute__((address_space(3))) v4i lds_v4i_t;
#else
#define USE_ASYNC_LDS 0
#endif

// ---------------------------------------------------------------------------
// WMMA fragment load, per CDNA5 ISA 16-bit A/B layout:
// lane L (0..15) -> row (M or N) = L, K chunks [kh,kh+8) and [16+kh,16+kh+8)
// lane L (16..31) -> row L-16, kh = 8.
// Both A ([M,K] row-major) and B (weights stored [N,K] row-major) use this.
// ---------------------------------------------------------------------------
static __device__ inline v16h load_frag(const _Float16* __restrict__ base, int ld,
                                        int row0, int k0, int lane) {
  int r  = row0 + (lane & 15);
  int kh = (lane >> 4) << 3;
  const _Float16* p = base + (long)r * ld + k0 + kh;
  h8 lo = *(const h8*)p;
  h8 hi = *(const h8*)(p + 16);
  v16h f;
#pragma unroll
  for (int i = 0; i < 8; ++i) { f[i] = lo[i]; f[8 + i] = hi[i]; }
  return f;
}

static __device__ inline v8f wmma_step(v16h a, v16h b, v8f c) {
  return __builtin_amdgcn_wmma_f32_16x16x32_f16(false, a, false, b, (short)0, c,
                                                false, false);
}

// ---------------------------------------------------------------------------
// Generic WMMA GEMM:  C[M,N] = scale * (A[M,K] @ W[N,K]^T) + bias[N]
// A, W are f16; accumulate f32; optional f32 and/or f16 stores.
// blockIdx.y = "head", with per-head element strides sAh/sBh/sCh.
// Each wave computes a 32x32 C tile (4 accumulators; A/B fragments reused
// across two WMMAs each -> 16 FLOP/B and 4 independent WMMA chains).
// Requires M,N multiples of 32; K multiple of 32.
// ---------------------------------------------------------------------------
__global__ __launch_bounds__(256) void gemm_wmma_f16(
    const _Float16* __restrict__ A, int lda, long sAh,
    const _Float16* __restrict__ Bm, int ldb, long sBh,
    const float* __restrict__ bias,
    float* __restrict__ Cf, _Float16* __restrict__ Ch, int ldc, long sCh,
    int M, int N, int K, float scale)
{
  const int head = blockIdx.y;
  A  += (long)head * sAh;
  Bm += (long)head * sBh;

  const int wave = threadIdx.x >> 5;
  const int lane = threadIdx.x & 31;
  const int tiles_n = N >> 5;
  const int tiles_m = M >> 5;
  const long tid = (long)blockIdx.x * (blockDim.x >> 5) + wave;
  if (tid >= (long)tiles_m * tiles_n) return;   // wave-uniform exit (EXEC stays all-1)

  const int tm = (int)(tid / tiles_n) << 5;
  const int tn = (int)(tid % tiles_n) << 5;

  v8f acc00 = {}, acc01 = {}, acc10 = {}, acc11 = {};
  for (int k0 = 0; k0 < K; k0 += 32) {
    if (k0 + 32 < K) {  // gfx1250 global_prefetch_b8 for the next K chunk
      __builtin_prefetch(A  + (long)(tm + (lane & 15)) * lda + k0 + 32, 0, 3);
      __builtin_prefetch(Bm + (long)(tn + (lane & 15)) * ldb + k0 + 32, 0, 3);
    }
    v16h a0 = load_frag(A,  lda, tm,      k0, lane);
    v16h a1 = load_frag(A,  lda, tm + 16, k0, lane);
    v16h b0 = load_frag(Bm, ldb, tn,      k0, lane);
    v16h b1 = load_frag(Bm, ldb, tn + 16, k0, lane);
    acc00 = wmma_step(a0, b0, acc00);
    acc01 = wmma_step(a0, b1, acc01);
    acc10 = wmma_step(a1, b0, acc10);
    acc11 = wmma_step(a1, b1, acc11);
  }

  if (Cf) Cf += (long)head * sCh;
  if (Ch) Ch += (long)head * sCh;

  // C/D layout per 16x16 sub-tile: lane 0-15 -> N=lane, M = vg (+8 for 16-31)
  const int nlo = lane & 15;
  const int mhi = (lane >> 4) << 3;
  v8f accs[4] = {acc00, acc01, acc10, acc11};
#pragma unroll
  for (int s = 0; s < 4; ++s) {
    const int ti = (s >> 1) << 4;   // 0,0,16,16
    const int tj = (s & 1) << 4;    // 0,16,0,16
    const int n  = tn + tj + nlo;
    const float bv = bias ? bias[n] : 0.0f;
#pragma unroll
    for (int vg = 0; vg < 8; ++vg) {
      const int m = tm + ti + mhi + vg;
      const float x = accs[s][vg] * scale + bv;
      const long off = (long)m * ldc + n;
      if (Cf) Cf[off] = x;
      if (Ch) Ch[off] = (_Float16)x;
    }
  }
}

// ---------------------------------------------------------------------------
// Stage 1: time-encoding MLP + embedding gather/relu/sum + diffusion mix.
// One block per (b,v); thread = feature d. Writes V-major f16 ve/gen1 for the
// GEMM stages, and copies noise tensors straight into d_out.
// ---------------------------------------------------------------------------
__global__ __launch_bounds__(256) void embed_gen1_kernel(
    const int* __restrict__ seqs, const float* __restrict__ tstep,
    const int* __restrict__ dt, const float* __restrict__ nn,
    const float* __restrict__ pn, const float* __restrict__ emb,
    const float* __restrict__ alph,
    const float* __restrict__ tl_w, const float* __restrict__ tl_b,
    const float* __restrict__ tu_w, const float* __restrict__ tu_b,
    _Float16* __restrict__ ve16, _Float16* __restrict__ gen1_16,
    float* __restrict__ out_nn, float* __restrict__ out_pn)
{
  const int bv = blockIdx.x;
  const int b = bv / V, v = bv % V;
  const int d = threadIdx.x;

  __shared__ float hj[64];
  const float s = tstep[bv] * (1.0f / 180.0f);
  if (d < 64) { const float z = s * tl_w[d] + tl_b[d]; hj[d] = 1.0f - tanhf(z * z); }
  __syncthreads();

  float acc = tu_b[d];
#pragma unroll 8
  for (int j = 0; j < 64; ++j) acc += hj[j] * tu_w[d * 64 + j];

  const int* srow = seqs + (long)bv * Cc;
  float vesum = acc;
#pragma unroll 4
  for (int c = 0; c < Cc; ++c) {
    const float e = emb[(long)srow[c] * D + d];
    vesum += e > 0.0f ? e : 0.0f;
  }

  const float al  = alph[dt[b]];
  const long  fi  = (long)bv * D + d;
  const float nnv = nn[fi], pnv = pn[fi];
  const float ven = vesum * sqrtf(al) + nnv * sqrtf(1.0f - al);
  const float g1  = vesum + ven - pnv;

  const long vmaj = ((long)v * B + b) * D + d;   // [V,B,D]
  ve16[vmaj]   = (_Float16)vesum;
  gen1_16[vmaj] = (_Float16)g1;
  out_nn[fi] = nnv;
  out_pn[fi] = pnv;
}

// cumprod(1-betas) -> alph[T]; trivial serial prefix (1000 iters)
__global__ void cumprod_alpha_kernel(const float* __restrict__ betas, float* __restrict__ alph) {
  if (threadIdx.x == 0 && blockIdx.x == 0) {
    float p = 1.0f;
    for (int i = 0; i < T; ++i) { p *= (1.0f - betas[i]); alph[i] = p; }
  }
}

__global__ void f32_to_f16_kernel(const float* __restrict__ x, _Float16* __restrict__ y, long n) {
  const long i = (long)blockIdx.x * blockDim.x + threadIdx.x;
  if (i < n) y[i] = (_Float16)x[i];
}

__global__ void add_bias_kernel(const float* __restrict__ a, const float* __restrict__ b,
                                float* __restrict__ y, int n) {
  const int i = blockIdx.x * blockDim.x + threadIdx.x;
  if (i < n) y[i] = a[i] + b[i];
}

__global__ void fill_h16_kernel(_Float16* __restrict__ p, float val, long n) {
  const long i = (long)blockIdx.x * blockDim.x + threadIdx.x;
  if (i < n) p[i] = (_Float16)val;
}

// Row softmax over 256 cols (attention probabilities), f32 in / f16 out.
__global__ __launch_bounds__(256) void softmax_rows_kernel(
    const float* __restrict__ S, _Float16* __restrict__ P)
{
  const long row = blockIdx.x;
  const int t = threadIdx.x;
  __shared__ float red[256];
  const float x = S[row * 256 + t];
  red[t] = x; __syncthreads();
  for (int off = 128; off; off >>= 1) { if (t < off) red[t] = fmaxf(red[t], red[t + off]); __syncthreads(); }
  const float mx = red[0]; __syncthreads();
  const float e = __expf(x - mx);
  red[t] = e; __syncthreads();
  for (int off = 128; off; off >>= 1) { if (t < off) red[t] += red[t + off]; __syncthreads(); }
  P[row * 256 + t] = (_Float16)(e / red[0]);
}

// LDS-tiled transpose of the V-projection slice: vT[d, b] = qkv[b, 512 + d]
__global__ __launch_bounds__(256) void transpose_v_kernel(
    const _Float16* __restrict__ qkv, _Float16* __restrict__ vT)
{
  __shared__ _Float16 tile[32][33];
  const int bx = blockIdx.x * 32;            // d
  const int by = blockIdx.y * 32;            // b
  const int tx = threadIdx.x & 31, ty = threadIdx.x >> 5;
  for (int i = ty; i < 32; i += 8)
    tile[i][tx] = qkv[(long)(by + i) * (3 * D) + 2 * D + bx + tx];
  __syncthreads();
  for (int i = ty; i < 32; i += 8)
    vT[(long)(bx + i) * B + by + tx] = tile[tx][i];
}

static __device__ inline float sigf(float x) { return 1.0f / (1.0f + __expf(-x)); }

// ---------------------------------------------------------------------------
// LSTM recurrence: 50 independent LSTMs (one per block), 256 serial steps.
// xg (precomputed by WMMA GEMM, f16 [V, B, 4H]) + h @ w_hh^T each step.
// The next step's 2KB xg row is pipelined into LDS with the gfx1250 async
// global->LDS engine (ASYNCcnt) while the current step computes.
// ---------------------------------------------------------------------------
__global__ __launch_bounds__(1024) void lstm_rec_kernel(
    const _Float16* __restrict__ xg, const _Float16* __restrict__ whh16,
    float* __restrict__ hout)
{
  const int v = blockIdx.x;
  const int j = threadIdx.x;            // gate index 0..1023
  __shared__ float h[H], c[H], gates[4 * H];
#if USE_ASYNC_LDS
  __shared__ alignas(16) _Float16 xrow[2][4 * H];   // double-buffered gate rows
  _Float16* xg_nc = const_cast<_Float16*>(xg);
#endif
  if (j < H) { h[j] = 0.0f; c[j] = 0.0f; }

#if USE_ASYNC_LDS
  // prime buffer 0 with step 0's row (128 lanes x 16B)
  if (j < 128) {
    __builtin_amdgcn_global_load_async_to_lds_b128(
        (glob_v4i_t*)(xg_nc + (long)v * B * 4 * H + j * 8),
        (lds_v4i_t*)(&xrow[0][j * 8]), 0, 0);
    __builtin_amdgcn_s_wait_asynccnt(0);
  }
#endif
  __syncthreads();

  const h8* wr8 = (const h8*)(whh16 + (long)j * H);
  for (int b = 0; b < B; ++b) {
#if USE_ASYNC_LDS
    if (b + 1 < B && j < 128) {   // kick off next row while this step computes
      __builtin_amdgcn_global_load_async_to_lds_b128(
          (glob_v4i_t*)(xg_nc + ((long)v * B + b + 1) * 4 * H + j * 8),
          (lds_v4i_t*)(&xrow[(b + 1) & 1][j * 8]), 0, 0);
    }
    float acc = (float)xrow[b & 1][j];
#else
    float acc = (float)xg[(((long)v * B + b) * 4 * H) + j];
#endif
#pragma unroll 4
    for (int k8 = 0; k8 < H / 8; ++k8) {
      const h8 w = wr8[k8];
      const int k = k8 * 8;
      acc += (float)w[0] * h[k]     + (float)w[1] * h[k + 1]
           + (float)w[2] * h[k + 2] + (float)w[3] * h[k + 3]
           + (float)w[4] * h[k + 4] + (float)w[5] * h[k + 5]
           + (float)w[6] * h[k + 6] + (float)w[7] * h[k + 7];
    }
    gates[j] = acc;
    __syncthreads();
    if (j < H) {
      const float ig = gates[j], fg = gates[H + j], gg = gates[2 * H + j], og = gates[3 * H + j];
      const float cn = sigf(fg) * c[j] + sigf(ig) * tanhf(gg);
      c[j] = cn;
      h[j] = sigf(og) * tanhf(cn);
    }
#if USE_ASYNC_LDS
    if (j < 128) __builtin_amdgcn_s_wait_asynccnt(0);  // next row landed
#endif
    __syncthreads();
  }
  if (j < H) hout[(long)v * H + j] = h[j];
}

// Classifier MLP (tiny): relu(h@c1^T+b1) -> relu(@c2^T+b2) -> softmax(@co^T+bo)
__global__ __launch_bounds__(1024) void classify_kernel(
    const float* __restrict__ hv, const float* __restrict__ hg,
    const float* __restrict__ c1w, const float* __restrict__ c1b,
    const float* __restrict__ c2w, const float* __restrict__ c2b,
    const float* __restrict__ cow, const float* __restrict__ cob,
    float* __restrict__ res_v)     // [2, V, 2]
{
  const int v = blockIdx.x, br = blockIdx.y;
  const float* h0 = (br ? hg : hv) + (long)v * H;
  __shared__ float hin[H];
  __shared__ float h1[4 * H];
  __shared__ float r0s[32], r1s[32];
  const int j = threadIdx.x;

  if (j < H) hin[j] = h0[j];
  __syncthreads();

  float a = c1b[j];
#pragma unroll 4
  for (int k = 0; k < H; ++k) a += hin[k] * c1w[(long)j * H + k];
  h1[j] = a > 0.0f ? a : 0.0f;
  __syncthreads();

  float b2 = c2b[j];
  const float* w2 = c2w + (long)j * (4 * H);
#pragma unroll 4
  for (int k = 0; k < 4 * H; ++k) b2 += h1[k] * w2[k];
  const float h2 = b2 > 0.0f ? b2 : 0.0f;

  float p0 = h2 * cow[j];
  float p1 = h2 * cow[4 * H + j];
  for (int off = 16; off; off >>= 1) {
    p0 += __shfl_down(p0, off, 32);
    p1 += __shfl_down(p1, off, 32);
  }
  const int wv = j >> 5, ln = j & 31;
  if (ln == 0) { r0s[wv] = p0; r1s[wv] = p1; }
  __syncthreads();
  if (j == 0) {
    float l0 = cob[0], l1 = cob[1];
    for (int w = 0; w < 32; ++w) { l0 += r0s[w]; l1 += r1s[w]; }
    const float m = fmaxf(l0, l1);
    const float e0 = __expf(l0 - m), e1 = __expf(l1 - m), inv = 1.0f / (e0 + e1);
    res_v[((long)br * V + v) * 2 + 0] = e0 * inv;
    res_v[((long)br * V + v) * 2 + 1] = e1 * inv;
  }
}

// Broadcast per-v results to [B,V,2] outputs + last-column slices.
__global__ __launch_bounds__(256) void scatter_out_kernel(
    const float* __restrict__ res_v, float* __restrict__ out)
{
  const long i = (long)blockIdx.x * blockDim.x + threadIdx.x;
  if (i >= (long)B * V) return;
  const int b = (int)(i / V), v = (int)(i % V);
  const float r0 = res_v[v * 2 + 0],            r1 = res_v[v * 2 + 1];
  const float g0 = res_v[(V + v) * 2 + 0],      g1 = res_v[(V + v) * 2 + 1];
  out[i * 2 + 0] = r0;                 out[i * 2 + 1] = r1;
  out[OFF_RESG + i * 2 + 0] = g0;      out[OFF_RESG + i * 2 + 1] = g1;
  if (v == V - 1) {
    out[OFF_RL  + b * 2 + 0] = r0;  out[OFF_RL  + b * 2 + 1] = r1;
    out[OFF_RGL + b * 2 + 0] = g0;  out[OFF_RGL + b * 2 + 1] = g1;
  }
}

// ---------------------------------------------------------------------------
// Host orchestration
// ---------------------------------------------------------------------------
extern "C" void kernel_launch(void* const* d_in, const int* in_sizes, int n_in,
                              void* d_out, int out_size, void* d_ws, size_t ws_size,
                              hipStream_t stream) {
  const int*   input_seqs = (const int*)d_in[0];
  const float* seq_time   = (const float*)d_in[1];
  const int*   dt         = (const int*)d_in[2];
  const float* nn         = (const float*)d_in[3];
  const float* pn         = (const float*)d_in[4];
  const float* emb        = (const float*)d_in[5];
  const float* betas      = (const float*)d_in[6];
  const float* tl_w = (const float*)d_in[7];
  const float* tl_b = (const float*)d_in[8];
  const float* tu_w = (const float*)d_in[9];
  const float* tu_b = (const float*)d_in[10];
  const float* in_proj_w  = (const float*)d_in[11];
  const float* in_proj_b  = (const float*)d_in[12];
  const float* out_proj_w = (const float*)d_in[13];
  const float* out_proj_b = (const float*)d_in[14];
  const float* w_ih = (const float*)d_in[15];
  const float* w_hh = (const float*)d_in[16];
  const float* b_ih = (const float*)d_in[17];
  const float* b_hh = (const float*)d_in[18];
  const float* c1w  = (const float*)d_in[19];
  const float* c1b  = (const float*)d_in[20];
  const float* c2w  = (const float*)d_in[21];
  const float* c2b  = (const float*)d_in[22];
  const float* cow  = (const float*)d_in[23];
  const float* cob  = (const float*)d_in[24];
  float* out = (float*)d_out;

  // workspace carve-up (~78 MB)
  char* w = (char*)d_ws;
  auto alloc = [&](size_t bytes) -> void* {
    void* p = (void*)w; w += (bytes + 255) & ~(size_t)255; return p;
  };
  float*     alph      = (float*)alloc((size_t)T * 4);
  _Float16*  inproj16  = (_Float16*)alloc((size_t)3 * D * D * 2);
  _Float16*  outproj16 = (_Float16*)alloc((size_t)D * D * 2);
  _Float16*  wih16     = (_Float16*)alloc((size_t)4 * H * D * 2);
  _Float16*  whh16     = (_Float16*)alloc((size_t)4 * H * H * 2);
  float*     bias_ihhh = (float*)alloc((size_t)4 * H * 4);
  _Float16*  prev0     = (_Float16*)alloc((size_t)B * D * 2);
  _Float16*  ve16      = (_Float16*)alloc((size_t)V * B * D * 2);
  _Float16*  gen1_16   = (_Float16*)alloc((size_t)V * B * D * 2);
  _Float16*  gen2_16   = (_Float16*)alloc((size_t)V * B * D * 2);
  _Float16*  qkv16     = (_Float16*)alloc((size_t)B * 3 * D * 2);
  _Float16*  vT16      = (_Float16*)alloc((size_t)D * B * 2);
  float*     Sbuf      = (float*)alloc((size_t)NH * B * B * 4);
  _Float16*  P16       = (_Float16*)alloc((size_t)NH * B * B * 2);
  _Float16*  O16       = (_Float16*)alloc((size_t)B * D * 2);
  _Float16*  xgv16     = (_Float16*)alloc((size_t)V * B * 4 * H * 2);
  _Float16*  xgg16     = (_Float16*)alloc((size_t)V * B * 4 * H * 2);
  float*     hv        = (float*)alloc((size_t)V * H * 4);
  float*     hg        = (float*)alloc((size_t)V * H * 4);
  float*     res_v     = (float*)alloc((size_t)2 * V * 2 * 4);

  auto gemm = [&](const _Float16* A, int lda, long sAh,
                  const _Float16* Bm, int ldb, long sBh,
                  const float* bias, float* Cf, _Float16* Ch, int ldc, long sCh,
                  int M, int N, int K, int heads, float scale) {
    const int tiles = (M / 32) * (N / 32);    // 32x32 tile per wave
    dim3 g((tiles + 7) / 8, heads);
    gemm_wmma_f16<<<g, 256, 0, stream>>>(A, lda, sAh, Bm, ldb, sBh, bias,
                                         Cf, Ch, ldc, sCh, M, N, K, scale);
  };

  // --- prep ---
  cumprod_alpha_kernel<<<1, 32, 0, stream>>>(betas, alph);
  f32_to_f16_kernel<<<(3 * D * D + 255) / 256, 256, 0, stream>>>(in_proj_w,  inproj16,  (long)3 * D * D);
  f32_to_f16_kernel<<<(D * D + 255) / 256,     256, 0, stream>>>(out_proj_w, outproj16, (long)D * D);
  f32_to_f16_kernel<<<(4 * H * D + 255) / 256, 256, 0, stream>>>(w_ih, wih16, (long)4 * H * D);
  f32_to_f16_kernel<<<(4 * H * H + 255) / 256, 256, 0, stream>>>(w_hh, whh16, (long)4 * H * H);
  add_bias_kernel<<<(4 * H + 255) / 256, 256, 0, stream>>>(b_ih, b_hh, bias_ihhh, 4 * H);
  fill_h16_kernel<<<(B * D + 255) / 256, 256, 0, stream>>>(prev0, 10000.0f, (long)B * D);

  // --- stage 1: embeddings / time enc / diffusion mix (also copies noise outs)
  embed_gen1_kernel<<<B * V, 256, 0, stream>>>(
      input_seqs, seq_time, dt, nn, pn, emb, alph,
      tl_w, tl_b, tu_w, tu_b, ve16, gen1_16, out + OFF_NN, out + OFF_PN);

  // --- stage 2: 50 sequential MHA steps (gen2 scan) ---
  const float inv_sqrt_hd = 0.17677669529663687f;   // (D/NH)^-0.5 = 32^-0.5
  for (int v = 0; v < V; ++v) {
    const _Float16* prev = (v == 0) ? prev0 : gen2_16 + (long)(v - 1) * B * D;
    // Q = prev @ Wq^T + bq         -> qkv[:, 0:256]
    gemm(prev, D, 0, inproj16, D, 0, in_proj_b,
         nullptr, qkv16, 3 * D, 0, B, D, D, 1, 1.0f);
    // K,V = gen1[v] @ Wkv^T + bkv  -> qkv[:, 256:768]
    gemm(gen1_16 + (long)v * B * D, D, 0, inproj16 + (long)D * D, D, 0, in_proj_b + D,
         nullptr, qkv16 + D, 3 * D, 0, B, 2 * D, D, 1, 1.0f);
    // per-head scores: S[h] = Qh @ Kh^T * hd^-0.5
    gemm(qkv16, 3 * D, 32, qkv16 + D, 3 * D, 32, nullptr,
         Sbuf, nullptr, B, (long)B * B, B, B, 32, NH, inv_sqrt_hd);
    softmax_rows_kernel<<<NH * B, 256, 0, stream>>>(Sbuf, P16);
    transpose_v_kernel<<<dim3(8, 8), 256, 0, stream>>>(qkv16, vT16);
    // O[:, h*32:(h+1)*32] = P[h] @ Vh
    gemm(P16, B, (long)B * B, vT16, B, (long)32 * B, nullptr,
         nullptr, O16, D, 32, B, 32, B, NH, 1.0f);
    // gen2[v] = O @ Wo^T + bo
    gemm(O16, D, 0, outproj16, D, 0, out_proj_b,
         nullptr, gen2_16 + (long)v * B * D, D, 0, B, D, D, 1, 1.0f);
  }

  // --- stage 3: LSTM xg precompute (big WMMA GEMMs) + serial recurrences ---
  gemm(ve16,    D, 0, wih16, D, 0, bias_ihhh, nullptr, xgv16, 4 * H, 0, V * B, 4 * H, D, 1, 1.0f);
  gemm(gen2_16, D, 0, wih16, D, 0, bias_ihhh, nullptr, xgg16, 4 * H, 0, V * B, 4 * H, D, 1, 1.0f);
  lstm_rec_kernel<<<V, 1024, 0, stream>>>(xgv16, whh16, hv);
  lstm_rec_kernel<<<V, 1024, 0, stream>>>(xgg16, whh16, hg);

  // --- stage 4: classifier + scatter ---
  classify_kernel<<<dim3(V, 2), 1024, 0, stream>>>(hv, hg, c1w, c1b, c2w, c2b, cow, cob, res_v);
  scatter_out_kernel<<<(B * V + 255) / 256, 256, 0, stream>>>(res_v, out);
}
